// HGCN_31353261261174
// MI455X (gfx1250) — compile-verified, matrix operand order checked by reading, not run on hardware
//
#include <hip/hip_runtime.h>
#include <hip/hip_bf16.h>
#include <stdint.h>

// ---------------------------------------------------------------------------
// HGCN forward for MI455X (gfx1250), wave32.
//   - big GEMM (50000x1432 @ 1432x64) via v_wmma_f32_16x16x32_bf16
//   - W slices staged to LDS with global_load_async_to_lds_b128 (double
//     buffered, s_wait_asynccnt) when the toolchain exposes the builtin
//   - per-row hyperbolic chains fused into wave-per-row / thread-per-row
// ---------------------------------------------------------------------------

#define NROWS 50000
#define NEDGE 800000
#define DIN   1433
#define DINP  1440          // padded K (45 * 32)
#define NKB   45
#define DHID  64
#define DOUT  7
#define EPSF  1e-7f
#define MINN  1e-15f
#define MAXN  1e6f

typedef __bf16 v16bf __attribute__((ext_vector_type(16)));
typedef float  v8f   __attribute__((ext_vector_type(8)));
typedef int    v4i   __attribute__((ext_vector_type(4)));

#if __has_builtin(__builtin_amdgcn_global_load_async_to_lds_b128)
#define HGCN_ASYNC_LDS 1
typedef __attribute__((address_space(1))) v4i GV4;   // global int4 (prints as __device__)
typedef __attribute__((address_space(3))) v4i LV4;   // LDS int4 (prints as __shared__)
#else
#define HGCN_ASYNC_LDS 0
#endif

#if __has_builtin(__builtin_amdgcn_s_wait_asynccnt)
#define WAIT_ASYNC() __builtin_amdgcn_s_wait_asynccnt(0)
#else
#define WAIT_ASYNC() asm volatile("s_wait_asynccnt 0x0" ::: "memory")
#endif

// ---------------- helpers ----------------

__device__ inline float wredsum(float v) {
#pragma unroll
  for (int o = 16; o > 0; o >>= 1) v += __shfl_xor(v, o);
  return v;
}

// 8 fp32 -> bf16 fragment elements [e0..e0+7], accumulating sum of squares.
// GUARD: mask k==0 (proj_tan0) and k>=DIN (padding). Non-guarded path uses
// two dword-aligned 128b loads (steady-state K blocks 1..43).
template <bool GUARD>
__device__ inline void load8(const float* __restrict__ xrow, int kstart,
                             v16bf& f, int e0, float& ss) {
  if (GUARD) {
#pragma unroll
    for (int i = 0; i < 8; ++i) {
      int k = kstart + i;
      float v = (k >= 1 && k < DIN) ? xrow[k] : 0.f;
      ss += v * v;
      f[e0 + i] = (__bf16)v;
    }
  } else {
    float4 q0 = *(const float4*)(xrow + kstart);
    float4 q1 = *(const float4*)(xrow + kstart + 4);
    float vv[8] = {q0.x, q0.y, q0.z, q0.w, q1.x, q1.y, q1.z, q1.w};
#pragma unroll
    for (int i = 0; i < 8; ++i) {
      ss += vv[i] * vv[i];
      f[e0 + i] = (__bf16)vv[i];
    }
  }
}

// ---------------- prep: W1 -> bf16 (zero col0 + pad), hyp biases ----------------

__global__ void hgcn_prep(const float* __restrict__ W1, const float* __restrict__ b1,
                          const float* __restrict__ b2, __bf16* __restrict__ Wb,
                          float* __restrict__ hb1, float* __restrict__ hb2) {
  for (int i = threadIdx.x; i < DHID * DINP; i += 256) {
    int r = i / DINP, k = i - r * DINP;
    float val = (k >= 1 && k < DIN) ? W1[r * DIN + k] : 0.f;
    Wb[i] = (__bf16)val;
  }
  if (threadIdx.x == 0) {  // hb1 = proj(expmap0(proj_tan0(b1)))
    float n2 = 0.f;
    for (int d = 1; d < DHID; ++d) n2 += b1[d] * b1[d];
    float rn = fmaxf(sqrtf(n2), MINN);
    float sh = sinhf(rn);
    float sp = 0.f, tmp[DHID];
    for (int d = 1; d < DHID; ++d) { tmp[d] = sh * b1[d] / rn; sp += tmp[d] * tmp[d]; }
    hb1[0] = sqrtf(fmaxf(1.f + sp, EPSF));
    for (int d = 1; d < DHID; ++d) hb1[d] = tmp[d];
  }
  if (threadIdx.x == 1) {  // hb2
    float n2 = 0.f;
    for (int d = 1; d < DOUT; ++d) n2 += b2[d] * b2[d];
    float rn = fmaxf(sqrtf(n2), MINN);
    float sh = sinhf(rn);
    float sp = 0.f, tmp[DOUT];
    for (int d = 1; d < DOUT; ++d) { tmp[d] = sh * b2[d] / rn; sp += tmp[d] * tmp[d]; }
    hb2[0] = sqrtf(fmaxf(1.f + sp, EPSF));
    for (int d = 1; d < DOUT; ++d) hb2[d] = tmp[d];
  }
}

// ---------------- GEMM1: mv = scale(row) * (x_tan @ W1^T), WMMA bf16 ----------------
// 256 threads = 8 waves; each wave: 16 rows x 64 cols; LDS-staged W slices.

__global__ __launch_bounds__(256) void hgcn_gemm1(const float* __restrict__ x,
                                                  const __bf16* __restrict__ Wb,
                                                  float* __restrict__ mv) {
  __shared__ __bf16 WBs[2][DHID * 32];  // 2 x 4 KB K-slices of W (bf16)

  const int tid  = threadIdx.x;
  const int lane = tid & 31;
  const int wave = tid >> 5;
  const int half = lane >> 4;     // 0: K 0..7 / 16..23, 1: K 8..15 / 24..31
  const int mrow = lane & 15;     // A-matrix row within tile

  const int rowBase = blockIdx.x * 128 + wave * 16;
  const int rA = rowBase + mrow;
  const int rClamp = rA < NROWS ? rA : (NROWS - 1);
  const float* xrow = x + (size_t)rClamp * DIN;

  v8f acc[4] = {};
  float ss = 0.f;

  const int sr = tid >> 2;        // staging: row of W (0..63)
  const int sk = (tid & 3) * 8;   // staging: k offset (0,8,16,24)

  const int n = lane & 15;
  const int kbase = half * 8;

  // one K-step: B fragments from LDS slice, 4 column tiles of WMMA
  auto kstep = [&](const __bf16* wbuf, const v16bf& a) {
#pragma unroll
    for (int jt = 0; jt < 4; ++jt) {
      const __bf16* bp = &wbuf[(jt * 16 + n) * 32];
      v16bf b;
#pragma unroll
      for (int i = 0; i < 8; ++i) { b[i] = bp[kbase + i]; b[8 + i] = bp[16 + kbase + i]; }
      acc[jt] = __builtin_amdgcn_wmma_f32_16x16x32_bf16(
          false, a, false, b, (short)0, acc[jt], false, false);
    }
  };

#if HGCN_ASYNC_LDS
  // async (ASYNCcnt-tracked) staging, double-buffered
  auto stage = [&](int kb, int buf) {
    const __bf16* gp = Wb + (size_t)sr * DINP + kb * 32 + sk;
    __builtin_amdgcn_global_load_async_to_lds_b128(
        (GV4*)(uintptr_t)gp,
        (LV4*)(unsigned)(uintptr_t)&WBs[buf][sr * 32 + sk],
        0, 0);
  };

  stage(0, 0);

  {  // kb = 0 (guarded: k==0 masked)
    WAIT_ASYNC();
    __syncthreads();
    stage(1, 1);
    v16bf a;
    load8<true>(xrow, half * 8, a, 0, ss);
    load8<true>(xrow, 16 + half * 8, a, 8, ss);
    kstep(WBs[0], a);
  }
  for (int kb = 1; kb < NKB - 1; ++kb) {  // steady state, unguarded
    WAIT_ASYNC();
    __syncthreads();
    stage(kb + 1, (kb + 1) & 1);
    const int ko = kb * 32;
    __builtin_prefetch(xrow + ko + 128, 0, 0);
    v16bf a;
    load8<false>(xrow, ko + half * 8, a, 0, ss);
    load8<false>(xrow, ko + 16 + half * 8, a, 8, ss);
    kstep(WBs[kb & 1], a);
  }
  {  // kb = 44 (guarded tail: k >= 1433 masked)
    WAIT_ASYNC();
    __syncthreads();
    const int ko = (NKB - 1) * 32;
    v16bf a;
    load8<true>(xrow, ko + half * 8, a, 0, ss);
    load8<true>(xrow, ko + 16 + half * 8, a, 8, ss);
    kstep(WBs[(NKB - 1) & 1], a);
  }
#else
  // fallback: synchronous staging, single buffer, two barriers per slice
  for (int kb = 0; kb < NKB; ++kb) {
    const int ko = kb * 32;
    __syncthreads();
    *(uint4*)&WBs[0][sr * 32 + sk] = *(const uint4*)&Wb[(size_t)sr * DINP + ko + sk];
    __syncthreads();
    v16bf a;
    if (kb == 0 || kb == NKB - 1) {
      load8<true>(xrow, ko + half * 8, a, 0, ss);
      load8<true>(xrow, ko + 16 + half * 8, a, 8, ss);
    } else {
      load8<false>(xrow, ko + half * 8, a, 0, ss);
      load8<false>(xrow, ko + 16 + half * 8, a, 8, ss);
    }
    kstep(WBs[0], a);
  }
#endif

  // per-row scale: logmap0(proj(expmap0(proj_tan0(x)))) = scale * x_spatial
  float ssr = ss + __shfl_xor(ss, 16);           // full row sum of squares
  float r  = sqrtf(ssr);
  float th = fmaxf(r, MINN);
  float scale = acoshf(fmaxf(coshf(th), 1.f + EPSF)) / th;

#pragma unroll
  for (int g = 0; g < 8; ++g) {
    const int M = half * 8 + g;                  // C layout: lanes16-31 hold M=8..15
    const float sg = __shfl(scale, M);           // scale for row M lives at lane M
    const int rr = rowBase + M;
    if (rr < NROWS) {
#pragma unroll
      for (int jt = 0; jt < 4; ++jt)
        mv[(size_t)rr * DHID + jt * 16 + n] = sg * acc[jt][g];
    }
  }
}

// ---------------- stage A: hyp_linear tail (64-dim) -> logmap0, wave per row ----------------

__global__ __launch_bounds__(256) void hgcn_tail64(const float* __restrict__ mv,
                                                   const float* __restrict__ hb,
                                                   float* __restrict__ t1) {
  const int lane = threadIdx.x & 31;
  const int row = blockIdx.x * 8 + (threadIdx.x >> 5);
  if (row >= NROWS) return;

  const float ua = mv[(size_t)row * DHID + lane];
  const float ub = mv[(size_t)row * DHID + 32 + lane];
  const float hba = hb[lane], hbb = hb[32 + lane];
  const float hb0 = __shfl(hba, 0);
  const float hsa = lane ? hba : 0.f, hsb = hbb;

  // expmap0(mv) + proj -> point p
  const float xa = lane ? ua : 0.f, xb = ub;
  float xn = fmaxf(sqrtf(wredsum(xa * xa + xb * xb)), MINN);
  float sh = sinhf(xn);
  float psa = sh * xa / xn, psb = sh * xb / xn;
  float sp = wredsum(psa * psa + psb * psb);
  float p0 = sqrtf(fmaxf(1.f + sp, EPSF));

  // u = logmap0(hyp_bias)
  float hn = fmaxf(sqrtf(wredsum(hsa * hsa + hsb * hsb)), MINN);
  float thb = acoshf(fmaxf(hb0, 1.f + EPSF));
  float usa = thb * hsa / hn, usb = thb * hsb / hn;

  // ptransp0(p, u)
  float ynn = fmaxf(sqrtf(sp), MINN);
  float yna = psa / ynn, ynb = psb / ynn;
  float alpha = wredsum(yna * usa + ynb * usb);
  float om = 1.f - p0;
  float wsa = usa - alpha * om * yna, wsb = usb - alpha * om * ynb;

  // proj_tan(w, p)
  float ux = wredsum(psa * wsa + psb * wsb);
  float v0 = ux / fmaxf(p0, EPSF);

  // expmap(t=(v0,ws), p) + proj
  float S = wredsum(wsa * wsa + wsb * wsb);
  float md = S - v0 * v0;                  // minkowski_dot(t,t)
  float normu = fminf(sqrtf(fmaxf(md, EPSF)), MAXN);
  float th2 = fmaxf(normu, MINN);
  float c2 = coshf(th2), s2 = sinhf(th2) / th2;
  float ra = c2 * psa + s2 * wsa, rb = c2 * psb + s2 * wsb;
  float sn = wredsum(ra * ra + rb * rb);
  float r0 = sqrtf(fmaxf(1.f + sn, EPSF));

  // logmap0 (tangent input to aggregation)
  float yn2 = fmaxf(sqrtf(sn), MINN);
  float g = acoshf(fmaxf(r0, 1.f + EPSF)) / yn2;
  t1[(size_t)row * DHID + lane] = lane ? g * ra : 0.f;
  t1[(size_t)row * DHID + 32 + lane] = g * rb;
}

// ---------------- stage C: agg tail + hyp_act + logmap0 (64-dim), wave per row ----------------

__global__ __launch_bounds__(256) void hgcn_act64(const float* __restrict__ sup,
                                                  float* __restrict__ t2) {
  const int lane = threadIdx.x & 31;
  const int row = blockIdx.x * 8 + (threadIdx.x >> 5);
  if (row >= NROWS) return;

  const float sa = sup[(size_t)row * DHID + lane];
  const float sb = sup[(size_t)row * DHID + 32 + lane];
  const float za = lane ? sa : 0.f, zb = sb;

  // proj(expmap0(support))
  float xn = fmaxf(sqrtf(wredsum(za * za + zb * zb)), MINN);
  float sh = sinhf(xn);
  float psa = sh * za / xn, psb = sh * zb / xn;
  float sp = wredsum(psa * psa + psb * psb);
  float p0 = sqrtf(fmaxf(1.f + sp, EPSF));

  // hyp_act: relu(logmap0(p))
  float ln = fmaxf(sqrtf(sp), MINN);
  float g = acoshf(fmaxf(p0, 1.f + EPSF)) / ln;
  float xta = fmaxf(g * psa, 0.f), xtb = fmaxf(g * psb, 0.f);

  // proj(expmap0(xt))
  float mn = fmaxf(sqrtf(wredsum(xta * xta + xtb * xtb)), MINN);
  float sh2 = sinhf(mn);
  float qa = sh2 * xta / mn, qb = sh2 * xtb / mn;
  float q2 = wredsum(qa * qa + qb * qb);
  float q0 = sqrtf(fmaxf(1.f + q2, EPSF));

  // logmap0 (tangent input to layer-2 linear)
  float qn = fmaxf(sqrtf(q2), MINN);
  float gq = acoshf(fmaxf(q0, 1.f + EPSF)) / qn;
  t2[(size_t)row * DHID + lane] = lane ? gq * qa : 0.f;
  t2[(size_t)row * DHID + 32 + lane] = gq * qb;
}

// ---------------- aggregation scatters ----------------

__global__ __launch_bounds__(256) void hgcn_agg64(const int* __restrict__ rowi,
                                                  const int* __restrict__ coli,
                                                  const float* __restrict__ ew,
                                                  const float* __restrict__ t,
                                                  float* __restrict__ sup) {
  int tid = blockIdx.x * 256 + threadIdx.x;
  int e = tid >> 6, d = tid & 63;
  if (e >= NEDGE) return;
  int r = rowi[e], c = coli[e];
  atomicAdd(&sup[(size_t)r * DHID + d], ew[e] * t[(size_t)c * DHID + d]);
}

__global__ __launch_bounds__(256) void hgcn_agg7(const int* __restrict__ rowi,
                                                 const int* __restrict__ coli,
                                                 const float* __restrict__ ew,
                                                 const float* __restrict__ t,
                                                 float* __restrict__ sup) {
  int tid = blockIdx.x * 256 + threadIdx.x;
  int e = tid >> 3, d = tid & 7;
  if (e >= NEDGE || d >= DOUT) return;
  int r = rowi[e], c = coli[e];
  atomicAdd(&sup[(size_t)r * DOUT + d], ew[e] * t[(size_t)c * DOUT + d]);
}

// ---------------- layer 2: GEMV(7x64) + hyp_linear tail + logmap0, thread per row ----------------

__global__ __launch_bounds__(256) void hgcn_layer2(const float* __restrict__ t2,
                                                   const float* __restrict__ W2,
                                                   const float* __restrict__ hb2,
                                                   float* __restrict__ t3) {
  int row = blockIdx.x * 256 + threadIdx.x;
  if (row >= NROWS) return;
  float v[DHID];
  const float4* p4 = (const float4*)(t2 + (size_t)row * DHID);
#pragma unroll
  for (int i = 0; i < DHID / 4; ++i) {
    float4 q = p4[i];
    v[4 * i] = q.x; v[4 * i + 1] = q.y; v[4 * i + 2] = q.z; v[4 * i + 3] = q.w;
  }
  float m[DOUT];
#pragma unroll
  for (int j = 0; j < DOUT; ++j) {
    float s = 0.f;
    for (int k = 0; k < DHID; ++k) s += v[k] * W2[j * DHID + k];
    m[j] = s;
  }
  float hb[DOUT];
  for (int d = 0; d < DOUT; ++d) hb[d] = hb2[d];

  // --- hyp_linear tail, 7-dim, scalar ---
  float n2 = 0.f;
  for (int d = 1; d < DOUT; ++d) n2 += m[d] * m[d];
  float xn = fmaxf(sqrtf(n2), MINN);
  float sh = sinhf(xn);
  float p[DOUT]; p[0] = 0.f;
  float sp = 0.f;
  for (int d = 1; d < DOUT; ++d) { p[d] = sh * m[d] / xn; sp += p[d] * p[d]; }
  float p0 = sqrtf(fmaxf(1.f + sp, EPSF));

  float hn2 = 0.f;
  for (int d = 1; d < DOUT; ++d) hn2 += hb[d] * hb[d];
  float hn = fmaxf(sqrtf(hn2), MINN);
  float thb = acoshf(fmaxf(hb[0], 1.f + EPSF));
  float us[DOUT]; us[0] = 0.f;
  for (int d = 1; d < DOUT; ++d) us[d] = thb * hb[d] / hn;

  float ynn = fmaxf(sqrtf(sp), MINN);
  float alpha = 0.f;
  for (int d = 1; d < DOUT; ++d) alpha += (p[d] / ynn) * us[d];
  float om = 1.f - p0;
  float wv[DOUT]; wv[0] = 0.f;
  for (int d = 1; d < DOUT; ++d) wv[d] = us[d] - alpha * om * (p[d] / ynn);

  float uxp = 0.f;
  for (int d = 1; d < DOUT; ++d) uxp += p[d] * wv[d];
  float v0 = uxp / fmaxf(p0, EPSF);

  float S = 0.f;
  for (int d = 1; d < DOUT; ++d) S += wv[d] * wv[d];
  float md = S - v0 * v0;
  float normu = fminf(sqrtf(fmaxf(md, EPSF)), MAXN);
  float th2 = fmaxf(normu, MINN);
  float c2 = coshf(th2), s2 = sinhf(th2) / th2;
  float rs[DOUT];
  float sn = 0.f;
  for (int d = 1; d < DOUT; ++d) { rs[d] = c2 * p[d] + s2 * wv[d]; sn += rs[d] * rs[d]; }
  float r0 = sqrtf(fmaxf(1.f + sn, EPSF));

  float yn2 = fmaxf(sqrtf(sn), MINN);
  float g = acoshf(fmaxf(r0, 1.f + EPSF)) / yn2;
  t3[(size_t)row * DOUT + 0] = 0.f;
  for (int d = 1; d < DOUT; ++d) t3[(size_t)row * DOUT + d] = g * rs[d];
}

// ---------------- final: agg tail + hyp_act + proj_tan0(logmap0), thread per row ----------------

__global__ __launch_bounds__(256) void hgcn_final(const float* __restrict__ sup,
                                                  float* __restrict__ out) {
  int row = blockIdx.x * 256 + threadIdx.x;
  if (row >= NROWS) return;
  float s[DOUT];
  for (int d = 0; d < DOUT; ++d) s[d] = sup[(size_t)row * DOUT + d];

  float n2 = 0.f;
  for (int d = 1; d < DOUT; ++d) n2 += s[d] * s[d];
  float xn = fmaxf(sqrtf(n2), MINN);
  float sh = sinhf(xn);
  float p[DOUT];
  float sp = 0.f;
  for (int d = 1; d < DOUT; ++d) { p[d] = sh * s[d] / xn; sp += p[d] * p[d]; }
  float p0 = sqrtf(fmaxf(1.f + sp, EPSF));

  float ln = fmaxf(sqrtf(sp), MINN);
  float g = acoshf(fmaxf(p0, 1.f + EPSF)) / ln;
  float xt[DOUT];
  float m2 = 0.f;
  for (int d = 1; d < DOUT; ++d) { xt[d] = fmaxf(g * p[d], 0.f); m2 += xt[d] * xt[d]; }

  float mn = fmaxf(sqrtf(m2), MINN);
  float sh2 = sinhf(mn);
  float q[DOUT];
  float q2 = 0.f;
  for (int d = 1; d < DOUT; ++d) { q[d] = sh2 * xt[d] / mn; q2 += q[d] * q[d]; }
  float q0 = sqrtf(fmaxf(1.f + q2, EPSF));

  float qn = fmaxf(sqrtf(q2), MINN);
  float gq = acoshf(fmaxf(q0, 1.f + EPSF)) / qn;
  out[(size_t)row * DOUT + 0] = 0.f;
  for (int d = 1; d < DOUT; ++d) out[(size_t)row * DOUT + d] = gq * q[d];
}

// ---------------- launch ----------------

extern "C" void kernel_launch(void* const* d_in, const int* in_sizes, int n_in,
                              void* d_out, int out_size, void* d_ws, size_t ws_size,
                              hipStream_t stream) {
  const float* x  = (const float*)d_in[0];
  const int* rowi = (const int*)d_in[1];
  const int* coli = (const int*)d_in[2];
  const float* ew = (const float*)d_in[3];
  const float* W1 = (const float*)d_in[4];
  const float* b1 = (const float*)d_in[5];
  const float* W2 = (const float*)d_in[6];
  const float* b2 = (const float*)d_in[7];
  float* out = (float*)d_out;

  char* ws = (char*)d_ws;
  __bf16* Wb  = (__bf16*)(ws);                       // 64*1440 bf16 = 184320 B
  float* hb1  = (float*)(ws + 184320);               // 64 f
  float* hb2  = (float*)(ws + 184576);               // 7 f
  float* bufA = (float*)(ws + 262144);               // N*64 f (mv1, then t2)
  float* bufB = (float*)(ws + 262144 + 12800000);    // N*64 f (t1, then t3)
  float* bufC = (float*)(ws + 262144 + 25600000);    // N*64 f (support1, support2)

  hgcn_prep<<<1, 256, 0, stream>>>(W1, b1, b2, Wb, hb1, hb2);

  // Layer 1
  hgcn_gemm1<<<(NROWS + 127) / 128, 256, 0, stream>>>(x, Wb, bufA);
  hgcn_tail64<<<(NROWS + 7) / 8, 256, 0, stream>>>(bufA, hb1, bufB);
  (void)hipMemsetAsync(bufC, 0, (size_t)NROWS * DHID * sizeof(float), stream);
  hgcn_agg64<<<(NEDGE * 64) / 256, 256, 0, stream>>>(rowi, coli, ew, bufB, bufC);
  hgcn_act64<<<(NROWS + 7) / 8, 256, 0, stream>>>(bufC, bufA);

  // Layer 2
  hgcn_layer2<<<(NROWS + 255) / 256, 256, 0, stream>>>(bufA, W2, hb2, bufB);
  (void)hipMemsetAsync(bufC, 0, (size_t)NROWS * DOUT * sizeof(float), stream);
  hgcn_agg7<<<(NEDGE * 8) / 256, 256, 0, stream>>>(rowi, coli, ew, bufB, bufC);
  hgcn_final<<<(NROWS + 255) / 256, 256, 0, stream>>>(bufC, out);
}
